// PLM_72791105733237
// MI455X (gfx1250) — compile-verified, matrix operand order checked by reading, not run on hardware
//
#include <hip/hip_runtime.h>
#include <hip/hip_bf16.h>
#include <hip/hip_fp16.h>

typedef __attribute__((ext_vector_type(16))) _Float16 v16h;
typedef __attribute__((ext_vector_type(8)))  _Float16 v8h;
typedef __attribute__((ext_vector_type(4)))  _Float16 v4h;
typedef __attribute__((ext_vector_type(8)))  float    v8f;

#define NNODES 8192
#define KDIM   512
#define NEG_SLOPE 0.01f
#define L2EPS  1e-12f

// ---------------------------------------------------------------------------
// Kernel 1: row-wise L2 normalize f32 -> f16 (output into workspace).
// One block per row, 128 threads, 4 elements per thread (float4 loads).
// ---------------------------------------------------------------------------
__global__ __launch_bounds__(128) void PLM_l2norm_f16(
    const float* __restrict__ proto, _Float16* __restrict__ pn) {
  const int row = blockIdx.x;
  const int t   = threadIdx.x;                       // 0..127
  const float4 v = ((const float4*)(proto + (size_t)row * KDIM))[t];

  float ss = v.x * v.x + v.y * v.y + v.z * v.z + v.w * v.w;
  // wave32 reduction
  #pragma unroll
  for (int off = 16; off > 0; off >>= 1)
    ss += __shfl_down(ss, off, 32);

  __shared__ float red[4];
  if ((t & 31) == 0) red[t >> 5] = ss;
  __syncthreads();
  const float total = red[0] + red[1] + red[2] + red[3];
  const float scale = 1.0f / fmaxf(sqrtf(total), L2EPS);

  v4h h;
  h[0] = (_Float16)(v.x * scale);
  h[1] = (_Float16)(v.y * scale);
  h[2] = (_Float16)(v.z * scale);
  h[3] = (_Float16)(v.w * scale);
  *(v4h*)(pn + (size_t)row * KDIM + t * 4) = h;      // 8-byte store
}

// ---------------------------------------------------------------------------
// Fragment loader: A(16x32) and B(32x16) f16 WMMA layouts are identical with
// lane = row/col (lane&15) and k = (lane>>4)*8 + i for i<8 (VGPRs 0-3) and
// k = 16 + (lane>>4)*8 + (i-8) for i>=8 (VGPRs 4-7). Two contiguous 16-byte
// loads per lane from a row-major f16 matrix. Constant `k` becomes an
// immediate IOFFSET on the b128 loads after full unroll.
// ---------------------------------------------------------------------------
__device__ __forceinline__ v16h load_frag(const _Float16* __restrict__ base,
                                          int k) {
  union { v16h f; v8h h[2]; } u;
  u.h[0] = *(const v8h*)(base + k);        // k-range [k .. k+7]
  u.h[1] = *(const v8h*)(base + k + 16);   // k-range [k+16 .. k+23]
  return u.f;
}

// ---------------------------------------------------------------------------
// Kernel 2: sim = pn @ pn^T with leaky-ReLU epilogue.
// Block: 256 threads = 8 waves; block tile 128x128; wave tile 64(M) x 32(N).
// K loop fully unrolled: 16 steps of 32, 8 v_wmma_f32_16x16x32_f16 per step,
// 6 fragment loads per step off fixed base pointers (immediate offsets).
// ---------------------------------------------------------------------------
__global__ __launch_bounds__(256, 2) void PLM_gram_wmma(
    const _Float16* __restrict__ pn, float* __restrict__ out) {
  const int lane  = threadIdx.x & 31;
  const int wid   = threadIdx.x >> 5;        // 0..7 (wave32)
  const int waveM = wid & 1;
  const int waveN = wid >> 1;

  const int rowBase = blockIdx.y * 128 + waveM * 64;  // 4 M-tiles of 16
  const int colBase = blockIdx.x * 128 + waveN * 32;  // 2 N-tiles of 16
  const int lrow = lane & 15;
  const int kgrp = (lane >> 4) * 8;

  const _Float16* aptr[4];
  #pragma unroll
  for (int tm = 0; tm < 4; ++tm)
    aptr[tm] = pn + (size_t)(rowBase + tm * 16 + lrow) * KDIM + kgrp;
  const _Float16* bptr[2];
  #pragma unroll
  for (int tn = 0; tn < 2; ++tn)
    bptr[tn] = pn + (size_t)(colBase + tn * 16 + lrow) * KDIM + kgrp;

  v8f acc[4][2];
  #pragma unroll
  for (int tm = 0; tm < 4; ++tm)
    #pragma unroll
    for (int tn = 0; tn < 2; ++tn)
      acc[tm][tn] = (v8f){0.f, 0.f, 0.f, 0.f, 0.f, 0.f, 0.f, 0.f};

  #pragma unroll
  for (int k = 0; k < KDIM; k += 32) {
    v16h a[4], b[2];
    #pragma unroll
    for (int tm = 0; tm < 4; ++tm) a[tm] = load_frag(aptr[tm], k);
    #pragma unroll
    for (int tn = 0; tn < 2; ++tn) b[tn] = load_frag(bptr[tn], k);

    #pragma unroll
    for (int tm = 0; tm < 4; ++tm)
      #pragma unroll
      for (int tn = 0; tn < 2; ++tn)
        acc[tm][tn] = __builtin_amdgcn_wmma_f32_16x16x32_f16(
            /*neg_a=*/false, a[tm], /*neg_b=*/false, b[tn],
            /*c_mod=*/(short)0, acc[tm][tn],
            /*reuse_a=*/false, /*reuse_b=*/false);
  }

  // Epilogue: leaky-ReLU + store. C/D layout: VGPR r, lanes 0-15 -> M=r,
  // lanes 16-31 -> M=8+r, N = lane&15.
  const int rOff = (lane >> 4) * 8;
  #pragma unroll
  for (int tm = 0; tm < 4; ++tm) {
    #pragma unroll
    for (int tn = 0; tn < 2; ++tn) {
      const int col = colBase + tn * 16 + lrow;
      #pragma unroll
      for (int r = 0; r < 8; ++r) {
        float v = acc[tm][tn][r];
        v = (v >= 0.0f) ? v : NEG_SLOPE * v;
        const int row = rowBase + tm * 16 + rOff + r;
        out[(size_t)row * NNODES + col] = v;
      }
    }
  }
}

// ---------------------------------------------------------------------------
extern "C" void kernel_launch(void* const* d_in, const int* in_sizes, int n_in,
                              void* d_out, int out_size, void* d_ws, size_t ws_size,
                              hipStream_t stream) {
  (void)in_sizes; (void)n_in; (void)out_size; (void)ws_size;
  // inputs: x (unused by reference output), proto
  const float* proto = (const float*)d_in[1];
  float* out = (float*)d_out;
  _Float16* pn = (_Float16*)d_ws;   // 8192*512*2 = 8 MB scratch (L2-resident)

  PLM_l2norm_f16<<<NNODES, 128, 0, stream>>>(proto, pn);

  dim3 grid(NNODES / 128, NNODES / 128);   // 64 x 64 blocks
  PLM_gram_wmma<<<grid, 256, 0, stream>>>(pn, out);
}
